// BigBirdBlockAttention_77610059038810
// MI455X (gfx1250) — matrix operand order repaired and checked by reading, not compiled, not run on hardware
//
#include <hip/hip_runtime.h>
#include <hip/hip_bf16.h>

typedef __bf16 bf16;
typedef __attribute__((ext_vector_type(16))) __bf16 v16bf;
typedef __attribute__((ext_vector_type(4)))  __bf16 v4bf;
typedef __attribute__((ext_vector_type(8)))  float  v8f;
typedef __attribute__((ext_vector_type(4)))  unsigned int v4u;
typedef __attribute__((ext_vector_type(4)))  unsigned int u32x4;
typedef __attribute__((ext_vector_type(8)))  int i32x8;
typedef __attribute__((ext_vector_type(4)))  int i32x4;

#define WMMA_BF16(a,b,c) \
  __builtin_amdgcn_wmma_f32_16x16x32_bf16(false,(a),false,(b),(short)0,(c),false,false)

// ---- TDM 2D tile load: global (row-major, stride in elems) -> LDS packed.
// D# per ISA ch.8: group0 = {count=1, lds_addr, gaddr57, type=2},
// group1 = {data_size=2B, tensor dims 2^20 (no clip), tile d0 x d1, stride0}.
__device__ __forceinline__ void tdm_load_2d(unsigned lds_addr, const void* gptr,
                                            unsigned tile_d0, unsigned tile_d1,
                                            unsigned stride_elems) {
  const unsigned long long ga = (unsigned long long)(uintptr_t)gptr;
  u32x4 g0;
  g0[0] = 1u;                                    // count=1, no gather
  g0[1] = lds_addr;                              // LDS byte address
  g0[2] = (unsigned)ga;                          // global addr [31:0]
  g0[3] = (unsigned)((ga >> 32) & 0x01FFFFFFu) | (2u << 30);  // [56:32] | type=2
  const unsigned dim0 = 1u << 20, dim1 = 1u << 20;
  i32x8 g1;
  g1[0] = (int)(1u << 16);                       // wg_mask=0, data_size=1 (2B)
  g1[1] = (int)((dim0 & 0xFFFFu) << 16);         // dim0[15:0] << 16
  g1[2] = (int)((dim0 >> 16) | ((dim1 & 0xFFFFu) << 16));
  g1[3] = (int)((dim1 >> 16) | (tile_d0 << 16)); // tile_dim0
  g1[4] = (int)(tile_d1);                        // tile_dim1 (tile_dim2=0)
  g1[5] = (int)stride_elems;                     // tensor_dim0_stride[31:0]
  g1[6] = 0;
  g1[7] = 0;
  const i32x4 z4 = {0, 0, 0, 0};
  const i32x8 z8 = {0, 0, 0, 0, 0, 0, 0, 0};
  __builtin_amdgcn_tensor_load_to_lds(g0, g1, z4, z4, z8, 0);
}

__device__ __forceinline__ unsigned lds_addr_of(const void* p) {
  return (unsigned)(uintptr_t)p;  // generic shared addr: low 32 bits = LDS offset
}

// ---- fragment loaders (ISA 7.12.2 layouts) ------------------------------
__device__ __forceinline__ v16bf load_a_frag(const bf16* base, int row, int rs,
                                             int kc, int lane) {
  const int h = (lane >> 4) & 1;
  const int off = row * rs + kc + h * 8;
  union { v16bf v; v4u u[2]; } f;
  f.u[0] = *(const v4u*)(base + off);
  f.u[1] = *(const v4u*)(base + off + 16);
  return f.v;
}
__device__ __forceinline__ v16bf load_b_frag(const bf16* base, int row, int rs,
                                             int kc, int lane) {
  const int h = (lane >> 4) & 1;
  const int off = row * rs + kc + h * 8;
  union { v16bf v; v4u u[2]; } f;
  f.u[0] = *(const v4u*)(base + off);
  f.u[1] = *(const v4u*)(base + off + 16);
  return f.v;
}

// ---- conversions ---------------------------------------------------------
__global__ __launch_bounds__(256) void cvt_x_kernel(const float* __restrict__ x,
                                                    bf16* __restrict__ xb) {
  const int i = (blockIdx.x * 256 + threadIdx.x) * 4;
  const float4 v = *(const float4*)(x + i);
  v4bf o = { (bf16)v.x, (bf16)v.y, (bf16)v.z, (bf16)v.w };
  *(v4bf*)(xb + i) = o;
}

__global__ __launch_bounds__(256) void cvt_w_kernel(const float* __restrict__ Wq,
                                                    const float* __restrict__ Wk,
                                                    const float* __restrict__ Wv,
                                                    const float* __restrict__ Wo,
                                                    bf16* __restrict__ wt_base) {
  const float* W = (blockIdx.z == 0) ? Wq : (blockIdx.z == 1) ? Wk
                   : (blockIdx.z == 2) ? Wv : Wo;
  bf16* Wt = wt_base + (size_t)blockIdx.z * 1024 * 1024;
  __shared__ float tile[64][65];
  const int n0 = blockIdx.x * 64, k0 = blockIdx.y * 64;
  const int tc = threadIdx.x & 63, tr = threadIdx.x >> 6;
#pragma unroll
  for (int j = 0; j < 16; ++j) {
    const int row = tr + j * 4;
    tile[row][tc] = W[(size_t)(k0 + row) * 1024 + n0 + tc];
  }
  __syncthreads();
#pragma unroll
  for (int j = 0; j < 16; ++j) {
    const int row = tr + j * 4;
    Wt[(size_t)(n0 + row) * 1024 + k0 + tc] = (bf16)tile[tc][row];
  }
}

// ---- shared GEMM body: 64x64 tile/WG, TDM double-buffered LDS -----------
// MODE: 0=Q, 1=K, 2=V(transposed), 3=output projection (fp32 out)
template <int IS_PROJ>
__device__ __forceinline__ void gemm_body(
    const bf16* __restrict__ A, const bf16* __restrict__ Wt,
    const float* __restrict__ bias, bf16* __restrict__ obf,
    float* __restrict__ of32, int mode) {
  __shared__ __align__(16) bf16 As[2][64 * 64];
  __shared__ __align__(16) bf16 Bs[2][64 * 64];
  const int tid = threadIdx.x;
  const int lane = tid & 31, wave = tid >> 5;
  const int half = (lane >> 4) & 1, r = lane & 15;
  const int m0 = blockIdx.y * 64, n0 = blockIdx.x * 64;
  v8f c[4] = {};
  const int nk = 16;
  if (wave == 0) {
    tdm_load_2d(lds_addr_of(&As[0][0]), A  + (size_t)m0 * 1024, 64, 64, 1024);
    tdm_load_2d(lds_addr_of(&Bs[0][0]), Wt + (size_t)n0 * 1024, 64, 64, 1024);
  }
  for (int i = 0; i < nk; ++i) {
    const int cur = i & 1;
    if (wave == 0) {
      if (i + 1 < nk) {
        const int k1 = (i + 1) * 64;
        tdm_load_2d(lds_addr_of(&As[cur ^ 1][0]), A  + (size_t)m0 * 1024 + k1,
                    64, 64, 1024);
        tdm_load_2d(lds_addr_of(&Bs[cur ^ 1][0]), Wt + (size_t)n0 * 1024 + k1,
                    64, 64, 1024);
        __builtin_amdgcn_s_wait_tensorcnt(2);  // current pair complete
      } else {
        __builtin_amdgcn_s_wait_tensorcnt(0);
      }
    }
    __syncthreads();
#pragma unroll
    for (int kc = 0; kc < 64; kc += 32) {
      const v16bf a = load_a_frag(As[cur], wave * 16 + r, 64, kc, lane);
      v16bf bfr[4];
#pragma unroll
      for (int nt = 0; nt < 4; ++nt)
        bfr[nt] = load_b_frag(Bs[cur], nt * 16 + r, 64, kc, lane);
#pragma unroll
      for (int nt = 0; nt < 4; ++nt) c[nt] = WMMA_BF16(a, bfr[nt], c[nt]);
    }
    __syncthreads();
  }
#pragma unroll
  for (int nt = 0; nt < 4; ++nt) {
    const int n = n0 + nt * 16 + r;
    const float bn = bias[n];
    const int hh = n >> 6, d = n & 63;
#pragma unroll
    for (int j = 0; j < 8; ++j) {
      const int mrow = m0 + wave * 16 + j + 8 * half;
      const float v = c[nt][j] + bn;
      if (IS_PROJ) {
        of32[(size_t)mrow * 1024 + n] = v;
      } else {
        const int bb = mrow >> 11, s = mrow & 2047;
        if (mode == 2)  // V transposed: [b][h][d][s]
          obf[(((size_t)(bb * 16 + hh)) * 64 + d) * 2048 + s] = (bf16)v;
        else            // Q/K: [b][h][s][d]
          obf[(((size_t)(bb * 16 + hh)) * 2048 + s) * 64 + d] = (bf16)v;
      }
    }
  }
}

__global__ __launch_bounds__(128) void gemm_qkv_kernel(
    const bf16* __restrict__ Xb, const bf16* __restrict__ Wts,
    const float* __restrict__ bq, const float* __restrict__ bk,
    const float* __restrict__ bv,
    bf16* __restrict__ Qo, bf16* __restrict__ Ko, bf16* __restrict__ Vo) {
  const int mode = blockIdx.z;
  const bf16* Wt = Wts + (size_t)mode * 1024 * 1024;
  const float* bias = (mode == 0) ? bq : (mode == 1) ? bk : bv;
  bf16* out = (mode == 0) ? Qo : (mode == 1) ? Ko : Vo;
  gemm_body<0>(Xb, Wt, bias, out, nullptr, mode);
}

__global__ __launch_bounds__(128) void gemm_out_kernel(
    const bf16* __restrict__ Oa, const bf16* __restrict__ Wt,
    const float* __restrict__ bo, float* __restrict__ out) {
  gemm_body<1>(Oa, Wt, bo, nullptr, out, 3);
}

// ---- block-sparse flash attention: 1 WG per (qblock, h, b) --------------
__global__ __launch_bounds__(128) void attn_kernel(
    const bf16* __restrict__ Q, const bf16* __restrict__ K,
    const bf16* __restrict__ V, const unsigned char* __restrict__ mask,
    bf16* __restrict__ O) {
  const int qb = blockIdx.x, h = blockIdx.y, b = blockIdx.z;
  __shared__ __align__(16) bf16 Qs[64 * 64];
  __shared__ __align__(16) bf16 Ks[2][64 * 64];
  __shared__ __align__(16) bf16 Vs[2][64 * 64];
  __shared__ __align__(16) bf16 Ps[4][16 * 64];
  __shared__ int blist[32];
  __shared__ int bcount;

  const int tid = threadIdx.x;
  const int lane = tid & 31, wave = tid >> 5;
  const int half = (lane >> 4) & 1, r = lane & 15;

  const bf16* Qg = Q + (((size_t)(b * 16 + h)) * 2048 + qb * 64) * 64;
  const bf16* Kg = K + ((size_t)(b * 16 + h)) * 2048 * 64;
  const bf16* Vg = V + ((size_t)(b * 16 + h)) * 64 * 2048;
#pragma unroll
  for (int i = 0; i < 4; ++i) {
    const int t = tid * 8 + i * 1024;
    *(v4u*)(Qs + t) = *(const v4u*)(Qg + t);  // rows contiguous
  }
  if (tid == 0) {
    int cnt = 0;
    for (int kb = 0; kb < 32; ++kb)
      if (mask[(size_t)(qb * 64) * 2048 + kb * 64]) blist[cnt++] = kb;
    bcount = cnt;
  }
  __syncthreads();

  const v16bf qa0 = load_a_frag(Qs, wave * 16 + r, 64, 0, lane);
  const v16bf qa1 = load_a_frag(Qs, wave * 16 + r, 64, 32, lane);

  v8f acc[4] = {};
  float mr[8], lr[8];
#pragma unroll
  for (int j = 0; j < 8; ++j) { mr[j] = -3.0e38f; lr[j] = 0.0f; }

  const int nblocks = bcount;
  const float SCALE = 0.125f;  // HD^-0.5

  if (wave == 0) {  // prologue: stage first K/V block via TDM
    const int kb0 = blist[0];
    tdm_load_2d(lds_addr_of(&Ks[0][0]), Kg + (size_t)kb0 * 64 * 64, 64, 64, 64);
    tdm_load_2d(lds_addr_of(&Vs[0][0]), Vg + (size_t)kb0 * 64, 64, 64, 2048);
  }

  for (int it = 0; it < nblocks; ++it) {
    const int cur = it & 1;
    if (wave == 0) {
      if (it + 1 < nblocks) {
        const int kbn = blist[it + 1];
        tdm_load_2d(lds_addr_of(&Ks[cur ^ 1][0]), Kg + (size_t)kbn * 64 * 64,
                    64, 64, 64);
        tdm_load_2d(lds_addr_of(&Vs[cur ^ 1][0]), Vg + (size_t)kbn * 64,
                    64, 64, 2048);
        __builtin_amdgcn_s_wait_tensorcnt(2);
      } else {
        __builtin_amdgcn_s_wait_tensorcnt(0);
      }
    }
    __syncthreads();

    v8f sc[4] = {};
    {
      v16bf kf[4];
#pragma unroll
      for (int nt = 0; nt < 4; ++nt)
        kf[nt] = load_b_frag(Ks[cur], nt * 16 + r, 64, 0, lane);
#pragma unroll
      for (int nt = 0; nt < 4; ++nt) sc[nt] = WMMA_BF16(qa0, kf[nt], sc[nt]);
#pragma unroll
      for (int nt = 0; nt < 4; ++nt)
        kf[nt] = load_b_frag(Ks[cur], nt * 16 + r, 64, 32, lane);
#pragma unroll
      for (int nt = 0; nt < 4; ++nt) sc[nt] = WMMA_BF16(qa1, kf[nt], sc[nt]);
    }

#pragma unroll
    for (int j = 0; j < 8; ++j) {
      float mx = sc[0][j];
      mx = fmaxf(mx, sc[1][j]);
      mx = fmaxf(mx, sc[2][j]);
      mx = fmaxf(mx, sc[3][j]);
#pragma unroll
      for (int xm = 1; xm <= 8; xm <<= 1)
        mx = fmaxf(mx, __shfl_xor(mx, xm, 32));
      const float mnew = fmaxf(mr[j], mx * SCALE);
      const float corr = __expf(mr[j] - mnew);
      mr[j] = mnew;
      lr[j] *= corr;
#pragma unroll
      for (int nt = 0; nt < 4; ++nt) acc[nt][j] *= corr;
      float ps = 0.0f;
#pragma unroll
      for (int nt = 0; nt < 4; ++nt) {
        const float p = __expf(sc[nt][j] * SCALE - mnew);
        ps += p;
        Ps[wave][(j + 8 * half) * 64 + nt * 16 + r] = (bf16)p;
      }
#pragma unroll
      for (int xm = 1; xm <= 8; xm <<= 1) ps += __shfl_xor(ps, xm, 32);
      lr[j] += ps;
    }

    // P x V (same-wave LDS round-trip re-layouts P into A-fragments)
#pragma unroll
    for (int kc = 0; kc < 64; kc += 32) {
      const v16bf pa = load_a_frag(&Ps[wave][0], r, 64, kc, lane);
      v16bf vf[4];
#pragma unroll
      for (int nt = 0; nt < 4; ++nt)
        vf[nt] = load_b_frag(Vs[cur], nt * 16 + r, 64, kc, lane);
#pragma unroll
      for (int nt = 0; nt < 4; ++nt) acc[nt] = WMMA_BF16(pa, vf[nt], acc[nt]);
    }
    __syncthreads();
  }

#pragma unroll
  for (int j = 0; j < 8; ++j) {
    const float inv = 1.0f / lr[j];
    const int srow = qb * 64 + wave * 16 + j + 8 * half;
#pragma unroll
    for (int nt = 0; nt < 4; ++nt) {
      const int e = h * 64 + nt * 16 + r;
      O[((size_t)b * 2048 + srow) * 1024 + e] = (bf16)(acc[nt][j] * inv);
    }
  }
}

extern "C" void kernel_launch(void* const* d_in, const int* in_sizes, int n_in,
                              void* d_out, int out_size, void* d_ws,
                              size_t ws_size, hipStream_t stream) {
  const float* x  = (const float*)d_in[0];
  const float* Wq = (const float*)d_in[1];
  const float* bq = (const float*)d_in[2];
  const float* Wk = (const float*)d_in[3];
  const float* bk = (const float*)d_in[4];
  const float* Wv = (const float*)d_in[5];
  const float* bv = (const float*)d_in[6];
  const float* Wo = (const float*)d_in[7];
  const float* bo = (const float*)d_in[8];
  const unsigned char* mask = (const unsigned char*)d_in[9];
  float* out = (float*)d_out;

  char* ws = (char*)d_ws;
  bf16* xb = (bf16*)(ws);                       //  8 MB  x bf16
  bf16* wt = (bf16*)(ws + ((size_t)8  << 20));  //  8 MB  4x transposed weights bf16
  bf16* Qb = (bf16*)(ws + ((size_t)16 << 20));  //  8 MB  [b][h][s][d]
  bf16* Kb = (bf16*)(ws + ((size_t)24 << 20));  //  8 MB  [b][h][s][d]
  bf16* Vb = (bf16*)(ws + ((size_t)32 << 20));  //  8 MB  [b][h][d][s]
  bf16* Ob = (bf16*)(ws + ((size_t)40 << 20));  //  8 MB  attn out [b*s][e]

  cvt_x_kernel<<<4096, 256, 0, stream>>>(x, xb);
  cvt_w_kernel<<<dim3(16, 16, 4), 256, 0, stream>>>(Wq, Wk, Wv, Wo, wt);
  gemm_qkv_kernel<<<dim3(16, 64, 3), 128, 0, stream>>>(xb, wt, bq, bk, bv,
                                                       Qb, Kb, Vb);
  attn_kernel<<<dim3(32, 16, 2), 128, 0, stream>>>(Qb, Kb, Vb, mask, Ob);
  gemm_out_kernel<<<dim3(16, 64, 1), 128, 0, stream>>>(
      Ob, wt + (size_t)3 * 1024 * 1024, bo, out);
}
